// GATNET_11347303596564
// MI455X (gfx1250) — matrix operand order repaired and checked by reading, not compiled, not run on hardware
//
#include <hip/hip_runtime.h>

typedef __attribute__((ext_vector_type(16))) _Float16 v16h;
typedef __attribute__((ext_vector_type(8)))  float    v8f;

#define NEG_SLOPE 0.2f

// ---------------------------------------------------------------------------
// f32 -> f16 weight conversion (done once per launch; weights are tiny)
// ---------------------------------------------------------------------------
__global__ void f32_to_f16_kernel(const float* __restrict__ src,
                                  _Float16* __restrict__ dst, int n) {
    int i = blockIdx.x * blockDim.x + threadIdx.x;
    if (i < n) dst[i] = (_Float16)src[i];
}

__global__ void fill_kernel(float* __restrict__ p, long long n, float v) {
    long long i = (long long)blockIdx.x * blockDim.x + threadIdx.x;
    if (i < n) p[i] = v;
}

// ---------------------------------------------------------------------------
// WMMA GEMM:  C[M x NCOLS] = A[M x 128] (f32) * B[128 x NCOLS] (f16), f32 acc
// 256 threads = 8 wave32s; each wave computes a 16 x NCOLS strip.
// B is staged in LDS in WMMA-fragment order so each lane reads one v16h.
// ---------------------------------------------------------------------------
template <int NCOLS>
__global__ __launch_bounds__(256)
void gemm_wmma_kernel(const float* __restrict__ A,
                      const _Float16* __restrict__ Bh,
                      float* __restrict__ C, int M) {
    constexpr int K  = 128;
    constexpr int NT = NCOLS / 16;   // n-tiles per wave
    constexpr int KS = K / 32;       // k-steps of 32

    __shared__ alignas(32) _Float16 Bl[KS * NT * 32 * 16];

    const int tid  = threadIdx.x;
    const int lane = tid & 31;
    const int wave = tid >> 5;

    // Stage B into fragment-order LDS:
    // Bl[((ks*NT+nt)*32+l)*16+t] = B[k][col], k = ks*32 + (l>=16?8:0)+t+(t>=8?8:0),
    // col = nt*16 + (l&15)
    constexpr int TOT = K * NCOLS;
    for (int f = tid; f < TOT; f += 256) {
        int t  = f & 15;
        int l  = (f >> 4) & 31;
        int nt = (f >> 9) % NT;
        int ks = (f >> 9) / NT;
        int k   = ks * 32 + ((l >= 16) ? 8 : 0) + t + ((t >= 8) ? 8 : 0);
        int col = nt * 16 + (l & 15);
        Bl[f] = Bh[k * NCOLS + col];
    }
    __syncthreads();

    const int base = blockIdx.x * 128 + wave * 16;
    int arow_i = base + (lane & 15);
    if (arow_i > M - 1) arow_i = M - 1;            // clamp for tail block
    const float* __restrict__ arow = A + (size_t)arow_i * K;
    const int koff = (lane < 16) ? 0 : 8;

    v8f acc[NT];
#pragma unroll
    for (int i = 0; i < NT; ++i)
#pragma unroll
        for (int j = 0; j < 8; ++j) acc[i][j] = 0.f;

#pragma unroll
    for (int ks = 0; ks < KS; ++ks) {
        const int k0 = ks * 32;
        float4 p0 = *(const float4*)(arow + k0 + koff);
        float4 p1 = *(const float4*)(arow + k0 + koff + 4);
        float4 p2 = *(const float4*)(arow + k0 + 16 + koff);
        float4 p3 = *(const float4*)(arow + k0 + 16 + koff + 4);
        v16h a;
        a[0]=(_Float16)p0.x;  a[1]=(_Float16)p0.y;  a[2]=(_Float16)p0.z;  a[3]=(_Float16)p0.w;
        a[4]=(_Float16)p1.x;  a[5]=(_Float16)p1.y;  a[6]=(_Float16)p1.z;  a[7]=(_Float16)p1.w;
        a[8]=(_Float16)p2.x;  a[9]=(_Float16)p2.y;  a[10]=(_Float16)p2.z; a[11]=(_Float16)p2.w;
        a[12]=(_Float16)p3.x; a[13]=(_Float16)p3.y; a[14]=(_Float16)p3.z; a[15]=(_Float16)p3.w;
#pragma unroll
        for (int nt = 0; nt < NT; ++nt) {
            v16h b = *(const v16h*)&Bl[((ks * NT + nt) * 32 + lane) * 16];
            acc[nt] = __builtin_amdgcn_wmma_f32_16x16x32_f16(
                false, a, false, b, (short)0, acc[nt], false, false);
        }
    }

    // D layout: lanes 0-15 -> (M=r,    N=lane), lanes 16-31 -> (M=r+8, N=lane-16)
    const int ncol = lane & 15;
#pragma unroll
    for (int nt = 0; nt < NT; ++nt) {
#pragma unroll
        for (int r = 0; r < 8; ++r) {
            int orow = base + ((lane < 16) ? r : (r + 8));
            if (orow < M)
                C[(size_t)orow * NCOLS + nt * 16 + ncol] = acc[nt][r];
        }
    }
}

// ---------------------------------------------------------------------------
// Per-node attention scores: a_s[n,h] = <h[n,h,:], att_s[h,:]> (one wave/node)
// ---------------------------------------------------------------------------
template <int HCOLS>   // H*64
__global__ __launch_bounds__(256)
void scores_kernel(const float* __restrict__ h,
                   const float* __restrict__ att_s,
                   const float* __restrict__ att_d,
                   float* __restrict__ a_s, float* __restrict__ a_d, int Nn) {
    constexpr int H  = HCOLS / 64;
    constexpr int CT = HCOLS / 32;   // cols per lane
    constexpr int G  = 32 / H;       // lanes per head group
    const int lane = threadIdx.x & 31;
    const int wid  = blockIdx.x * (blockDim.x >> 5) + (threadIdx.x >> 5);
    if (wid >= Nn) return;
    const float* hr = h + (size_t)wid * HCOLS;
    const int c0 = lane * CT;
    float ps = 0.f, pd = 0.f;
#pragma unroll
    for (int i = 0; i < CT; ++i) {
        float v = hr[c0 + i];
        ps += v * att_s[c0 + i];
        pd += v * att_d[c0 + i];
    }
#pragma unroll
    for (int off = G >> 1; off > 0; off >>= 1) {
        ps += __shfl_xor(ps, off, 32);
        pd += __shfl_xor(pd, off, 32);
    }
    if ((lane & (G - 1)) == 0) {
        a_s[wid * H + lane / G] = ps;
        a_d[wid * H + lane / G] = pd;
    }
}

// ---------------------------------------------------------------------------
// Edge pass 1: segment max of leakyrelu(a_s[src]+a_d[dst]) per (dst, head)
// Self-loops are virtual: edge id >= E  =>  src = dst = id - E.
// ---------------------------------------------------------------------------
template <int H>
__global__ __launch_bounds__(256)
void edge_max_kernel(const long long* __restrict__ ei, int E, int Nn,
                     const float* __restrict__ a_s, const float* __restrict__ a_d,
                     float* __restrict__ m) {
    long long t = (long long)blockIdx.x * blockDim.x + threadIdx.x;
    if (t >= (long long)E + Nn) return;
    int src, dst;
    if (t < E) { src = (int)ei[t]; dst = (int)ei[E + t]; }
    else       { src = dst = (int)(t - E); }
#pragma unroll
    for (int hh = 0; hh < H; ++hh) {
        float e = a_s[src * H + hh] + a_d[dst * H + hh];
        e = (e > 0.f) ? e : NEG_SLOPE * e;
        atomicMax(&m[dst * H + hh], e);
    }
}

// ---------------------------------------------------------------------------
// Edge pass 2 (the big one): one wave per edge.
//   w = exp(lrelu(a_s[src]+a_d[dst]) - m[dst]);  s[dst]+=w;  acc[dst]+=w*h[src]
// Normalization is deferred to a per-node kernel (saves one full edge pass).
// ---------------------------------------------------------------------------
template <int H>
__global__ __launch_bounds__(256)
void edge_accum_kernel(const long long* __restrict__ ei, int E, int Nn,
                       const float* __restrict__ a_s, const float* __restrict__ a_d,
                       const float* __restrict__ m,
                       const float* __restrict__ h,
                       float* __restrict__ acc, float* __restrict__ s) {
    constexpr int COLS = H * 64;
    constexpr int CT   = COLS / 32;
    const int lane = threadIdx.x & 31;
    long long e = (long long)blockIdx.x * 8 + (threadIdx.x >> 5);
    if (e >= (long long)E + Nn) return;
    int src, dst;
    if (e < E) { src = (int)ei[e]; dst = (int)ei[E + e]; }
    else       { src = dst = (int)(e - E); }

    const int c0 = lane * CT;
    const int hh = c0 >> 6;  // head for this lane's columns
    float ev = a_s[src * H + hh] + a_d[dst * H + hh];
    ev = (ev > 0.f) ? ev : NEG_SLOPE * ev;
    float w = __expf(ev - m[dst * H + hh]);
    if ((lane & (32 / H - 1)) == 0) atomicAdd(&s[dst * H + hh], w);

    const float* hp = h   + (size_t)src * COLS + c0;
    float*       ap = acc + (size_t)dst * COLS + c0;
    if (CT == 4) {
        float4 hv = *(const float4*)hp;
        atomicAdd(ap + 0, w * hv.x);
        atomicAdd(ap + 1, w * hv.y);
        atomicAdd(ap + 2, w * hv.z);
        atomicAdd(ap + 3, w * hv.w);
    } else {
        float2 hv = *(const float2*)hp;
        atomicAdd(ap + 0, w * hv.x);
        atomicAdd(ap + 1, w * hv.y);
    }
}

// ---------------------------------------------------------------------------
// Normalize layer 1 in place: x1 = elu(acc/s + b1)    (H=2, 128 cols)
// ---------------------------------------------------------------------------
__global__ __launch_bounds__(256)
void norm1_kernel(float* __restrict__ acc, const float* __restrict__ s,
                  const float* __restrict__ b, int Nn) {
    long long t = (long long)blockIdx.x * blockDim.x + threadIdx.x;
    if (t >= (long long)Nn * 128) return;
    int n = (int)(t >> 7);
    int c = (int)(t & 127);
    float v = acc[t] / s[n * 2 + (c >> 6)] + b[c];
    acc[t] = (v > 0.f) ? v : (__expf(v) - 1.f);
}

// Normalize layer 2 in place (H=1, 64 cols; mean over 1 head == identity)
__global__ __launch_bounds__(256)
void norm2_kernel(float* __restrict__ out, const float* __restrict__ s,
                  const float* __restrict__ b, int Nn) {
    long long t = (long long)blockIdx.x * blockDim.x + threadIdx.x;
    if (t >= (long long)Nn * 64) return;
    int n = (int)(t >> 6);
    int c = (int)(t & 63);
    out[t] = out[t] / s[n] + b[c];
}

// ---------------------------------------------------------------------------
extern "C" void kernel_launch(void* const* d_in, const int* in_sizes, int n_in,
                              void* d_out, int out_size, void* d_ws, size_t ws_size,
                              hipStream_t stream) {
    const float*     x   = (const float*)d_in[0];
    const long long* ei  = (const long long*)d_in[1];
    const float*     W1  = (const float*)d_in[2];
    const float*     As1 = (const float*)d_in[3];
    const float*     Ad1 = (const float*)d_in[4];
    const float*     b1  = (const float*)d_in[5];
    const float*     W2  = (const float*)d_in[6];
    const float*     As2 = (const float*)d_in[7];
    const float*     Ad2 = (const float*)d_in[8];
    const float*     b2  = (const float*)d_in[9];

    const int N  = in_sizes[0] / 128;
    const int E  = in_sizes[1] / 2;
    const long long ET = (long long)E + N;  // edges incl. virtual self-loops

    // workspace layout (floats)
    float* ws = (float*)d_ws;
    size_t o = 0;
    float* h    = ws + o; o += (size_t)N * 128;   // h1, later reused for h2
    float* acc  = ws + o; o += (size_t)N * 128;   // layer-1 accum -> x1 (in place)
    float* as1  = ws + o; o += 2 * (size_t)N;
    float* ad1  = ws + o; o += 2 * (size_t)N;
    float* m1   = ws + o; o += 2 * (size_t)N;
    float* s1   = ws + o; o += 2 * (size_t)N;
    float* as2  = ws + o; o += (size_t)N;
    float* ad2  = ws + o; o += (size_t)N;
    float* m2   = ws + o; o += (size_t)N;
    float* s2   = ws + o; o += (size_t)N;
    _Float16* w1h = (_Float16*)(ws + o);
    _Float16* w2h = w1h + 128 * 128;
    float* outp = (float*)d_out;

    // weight conversion (f32 -> f16 for WMMA)
    f32_to_f16_kernel<<<(128 * 128 + 255) / 256, 256, 0, stream>>>(W1, w1h, 128 * 128);
    f32_to_f16_kernel<<<(128 * 64 + 255) / 256, 256, 0, stream>>>(W2, w2h, 128 * 64);

    // ---------------- layer 1 ----------------
    gemm_wmma_kernel<128><<<(N + 127) / 128, 256, 0, stream>>>(x, w1h, h, N);
    scores_kernel<128><<<(N + 7) / 8, 256, 0, stream>>>(h, As1, Ad1, as1, ad1, N);

    fill_kernel<<<(int)((2LL * N + 255) / 256), 256, 0, stream>>>(m1, 2LL * N, -1e30f);
    fill_kernel<<<(int)((2LL * N + 255) / 256), 256, 0, stream>>>(s1, 2LL * N, 0.f);
    fill_kernel<<<(int)(((long long)N * 128 + 255) / 256), 256, 0, stream>>>(acc, (long long)N * 128, 0.f);

    edge_max_kernel<2><<<(int)((ET + 255) / 256), 256, 0, stream>>>(ei, E, N, as1, ad1, m1);
    edge_accum_kernel<2><<<(int)((ET + 7) / 8), 256, 0, stream>>>(ei, E, N, as1, ad1, m1, h, acc, s1);
    norm1_kernel<<<(int)(((long long)N * 128 + 255) / 256), 256, 0, stream>>>(acc, s1, b1, N);

    // ---------------- layer 2 ----------------
    gemm_wmma_kernel<64><<<(N + 127) / 128, 256, 0, stream>>>(acc, w2h, h, N);
    scores_kernel<64><<<(N + 7) / 8, 256, 0, stream>>>(h, As2, Ad2, as2, ad2, N);

    fill_kernel<<<(int)(((long long)N + 255) / 256), 256, 0, stream>>>(m2, N, -1e30f);
    fill_kernel<<<(int)(((long long)N + 255) / 256), 256, 0, stream>>>(s2, N, 0.f);
    fill_kernel<<<(int)(((long long)N * 64 + 255) / 256), 256, 0, stream>>>(outp, (long long)N * 64, 0.f);

    edge_max_kernel<1><<<(int)((ET + 255) / 256), 256, 0, stream>>>(ei, E, N, as2, ad2, m2);
    edge_accum_kernel<1><<<(int)((ET + 7) / 8), 256, 0, stream>>>(ei, E, N, as2, ad2, m2, h, outp, s2);
    norm2_kernel<<<(int)(((long long)N * 64 + 255) / 256), 256, 0, stream>>>(outp, s2, b2, N);
}
